// ExpandFrame_7000796693058
// MI455X (gfx1250) — matrix operand order repaired and checked by reading, not compiled, not run on hardware
//
#include <hip/hip_runtime.h>
#include <hip/hip_bf16.h>

// Problem constants (static per reference)
#define B_     16
#define TTXT   1024   // T_text
#define DDIM   512    // D
#define TMEL   2049   // T_TOTAL + 1
#define MBLK   32     // m-rows per block (2 WMMA m-tiles share the B stream)
#define MBLOCKS 65    // ceil(2049/32)

typedef __attribute__((ext_vector_type(16))) __bf16 bf16x16;
typedef __attribute__((ext_vector_type(8)))  float  f32x8;

// ---------------------------------------------------------------------------
// Kernel 1: per-batch inclusive scan of duration -> centers c = e - 0.5*round(total)
// One block of 1024 threads per batch (32 waves).
// ---------------------------------------------------------------------------
__global__ __launch_bounds__(TTXT)
void centers_kernel(const float* __restrict__ dur, float* __restrict__ cc) {
    __shared__ float s[TTXT];
    const int b = blockIdx.x;
    const int t = threadIdx.x;
    s[t] = dur[(size_t)b * TTXT + t];
    // Hillis-Steele inclusive scan
    for (int off = 1; off < TTXT; off <<= 1) {
        __syncthreads();
        float v = (t >= off) ? s[t - off] : 0.0f;
        __syncthreads();
        s[t] += v;
    }
    __syncthreads();
    const float total = s[TTXT - 1];              // sum of durations
    cc[(size_t)b * TTXT + t] = s[t] - 0.5f * rintf(total);
}

// ---------------------------------------------------------------------------
// Kernel 2: enc [B,T,D] fp32  ->  encT [B,D,T] bf16 (LDS-tiled transpose)
// Block (32,8), each block does a 32x32 tile. encT stays L2-resident (16 MB).
// ---------------------------------------------------------------------------
__global__ __launch_bounds__(256)
void transpose_cvt_kernel(const float* __restrict__ enc, __bf16* __restrict__ encT) {
    __shared__ float tile[32][33];                // +1 pad kills bank conflicts
    const int b  = blockIdx.z;
    const int d0 = blockIdx.x * 32;
    const int t0 = blockIdx.y * 32;
    const int tx = threadIdx.x, ty = threadIdx.y;
#pragma unroll
    for (int i = 0; i < 4; ++i) {
        int tl = ty + 8 * i;
        tile[tl][tx] = enc[((size_t)b * TTXT + (t0 + tl)) * DDIM + d0 + tx];
    }
    __syncthreads();
#pragma unroll
    for (int i = 0; i < 4; ++i) {
        int dl = ty + 8 * i;
        encT[((size_t)b * DDIM + (d0 + dl)) * TTXT + t0 + tx] = (__bf16)tile[tx][dl];
    }
}

// ---------------------------------------------------------------------------
// Kernel 3: fused softmax + GEMM, 32 m-rows per block.
// Block = 256 threads (8 waves), one (batch, 32-row m-block) x full D=512.
// Phase A: 32 softmax rows (over T_text=1024) -> normalized bf16 A-plane in LDS (64 KB).
// Phase B: each wave owns 64 columns; per K-step: 4x ds_load_b128 (two A frags),
//          8x global_load_b128 (four B frags, shared by both m-tiles),
//          8x wmma_f32_16x16x32_bf16.  B L2-traffic halved vs 16-row blocks.
// ---------------------------------------------------------------------------
__global__ __launch_bounds__(256)
void gauss_attn_gemm_kernel(const float* __restrict__ cc,
                            const __bf16* __restrict__ encT,
                            float* __restrict__ out) {
    __shared__ __bf16 s_a[MBLK][TTXT];            // 64 KB normalized weights (A plane, row-major m x t)

    const int b    = blockIdx.y;
    const int m0   = blockIdx.x * MBLK;
    const int tid  = threadIdx.x;
    const int wave = tid >> 5;
    const int lane = tid & 31;

    const float* cbase = cc + (size_t)b * TTXT;   // 4 KB, L2-resident (re-read per block)

    // ---- Phase A: stable softmax over t for 32 m-rows (4 rows per wave) ----
    for (int r = wave; r < MBLK; r += 8) {
        const float m = (float)(m0 + r);          // mel frame index (rows >= TMEL are dead but finite)
        float lg[32];
        float lmax = -3.0e38f;
#pragma unroll
        for (int j = 0; j < 32; ++j) {
            float d = m - cbase[j * 32 + lane];
            float l = -0.1f * d * d;
            lg[j] = l;
            lmax  = fmaxf(lmax, l);
        }
#pragma unroll
        for (int off = 16; off > 0; off >>= 1)
            lmax = fmaxf(lmax, __shfl_xor(lmax, off, 32));
        float s = 0.0f;
#pragma unroll
        for (int j = 0; j < 32; ++j) {
            float e = __expf(lg[j] - lmax);
            lg[j] = e;
            s += e;
        }
#pragma unroll
        for (int off = 16; off > 0; off >>= 1)
            s += __shfl_xor(s, off, 32);
        const float inv = 1.0f / s;
#pragma unroll
        for (int j = 0; j < 32; ++j)
            s_a[r][j * 32 + lane] = (__bf16)(lg[j] * inv);
    }
    __syncthreads();

    // ---- Phase B: WMMA GEMM, out_tile[32 x 64] per wave ----
    const int h  = lane >> 4;                     // lane half
    const int am = lane & 15;                     // A row (M) / C column (N)
    const int d0 = wave * 64;

    f32x8 acc[2][4];
#pragma unroll
    for (int u = 0; u < 2; ++u)
#pragma unroll
        for (int f = 0; f < 4; ++f)
#pragma unroll
            for (int i = 0; i < 8; ++i) acc[u][f][i] = 0.0f;

    const __bf16* bbase = encT + ((size_t)b * DDIM + d0 + am) * TTXT;

    for (int kt = 0; kt < 32; ++kt) {
        const int t0 = kt * 32;

        // A fragments from LDS: lane holds row am (+16), K runs [8h,8h+8) and [16+8h,16+8h+8)
        union { uint4 q[2]; bf16x16 v; } af[2];
#pragma unroll
        for (int u = 0; u < 2; ++u) {
            const __bf16* ap = &s_a[u * 16 + am][t0 + h * 8];
            af[u].q[0] = *(const uint4*)(ap);         // ds_load_b128
            af[u].q[1] = *(const uint4*)(ap + 16);    // ds_load_b128
        }

        const int kh = t0 + (h << 3);             // lane half's K runs [8h,8h+8) and [16+8h,16+8h+8) for B
#pragma unroll
        for (int f = 0; f < 4; ++f) {
            union { uint4 q[2]; bf16x16 v; } bfr;
            const __bf16* bp = bbase + (size_t)f * 16 * TTXT + kh;
            bfr.q[0] = *(const uint4*)(bp);       // global_load_b128
            bfr.q[1] = *(const uint4*)(bp + 16);  // global_load_b128
            __builtin_prefetch(bp + 256, 0, 3);   // pull B stream ahead (SE scope)
            // both m-tiles consume the same B fragment -> halved L2 traffic
            acc[0][f] = __builtin_amdgcn_wmma_f32_16x16x32_bf16(
                false, af[0].v, false, bfr.v, (short)0, acc[0][f], false, false);
            acc[1][f] = __builtin_amdgcn_wmma_f32_16x16x32_bf16(
                false, af[1].v, false, bfr.v, (short)0, acc[1][f], false, false);
        }
    }

    // ---- store: C/D layout -> VGPR r holds (row r + 8h, col am) ----
    if (m0 + MBLK <= TMEL) {
        // fast path (64 of 65 blocks): every row valid, no per-element exec juggling
#pragma unroll
        for (int u = 0; u < 2; ++u) {
            float* obase = out + ((size_t)b * TMEL + m0 + u * 16) * DDIM + d0 + am;
#pragma unroll
            for (int f = 0; f < 4; ++f)
#pragma unroll
                for (int r = 0; r < 8; ++r)
                    obase[(size_t)(r + h * 8) * DDIM + f * 16] = acc[u][f][r];
        }
    } else {
        // ragged last block (m0 == 2048): guard each row
#pragma unroll
        for (int u = 0; u < 2; ++u) {
            float* obase = out + ((size_t)b * TMEL + m0 + u * 16) * DDIM + d0 + am;
#pragma unroll
            for (int f = 0; f < 4; ++f)
#pragma unroll
                for (int r = 0; r < 8; ++r)
                    if (m0 + u * 16 + r + h * 8 < TMEL)
                        obase[(size_t)(r + h * 8) * DDIM + f * 16] = acc[u][f][r];
        }
    }
}

// ---------------------------------------------------------------------------
extern "C" void kernel_launch(void* const* d_in, const int* in_sizes, int n_in,
                              void* d_out, int out_size, void* d_ws, size_t ws_size,
                              hipStream_t stream) {
    const float* enc = (const float*)d_in[0];     // [B, T_text, D] fp32
    const float* dur = (const float*)d_in[1];     // [B, T_text]   fp32
    // d_in[2] = t_mel (static 2049, unused)

    float*  out   = (float*)d_out;                        // [B, T_mel, D] fp32
    float*  c_ws  = (float*)d_ws;                         // 16*1024 fp32 = 64 KB
    __bf16* encT  = (__bf16*)((char*)d_ws + (size_t)B_ * TTXT * sizeof(float)); // 16 MB

    centers_kernel<<<dim3(B_), dim3(TTXT), 0, stream>>>(dur, c_ws);

    transpose_cvt_kernel<<<dim3(DDIM / 32, TTXT / 32, B_), dim3(32, 8), 0, stream>>>(enc, encT);

    gauss_attn_gemm_kernel<<<dim3(MBLOCKS, B_), dim3(256), 0, stream>>>(c_ws, encT, out);
}
